// LayerNormLayer_3161095929892
// MI455X (gfx1250) — matrix operand order repaired and checked by reading, not compiled, not run on hardware
//
#include <hip/hip_runtime.h>
#include <hip/hip_bf16.h>

// ---------------------------------------------------------------------------
// LN + MLP(relu) + dropout + residual on gfx1250.
// GEMMs: v_wmma_f32_16x16x32_bf16, LDS double-buffered via
// global_load_async_to_lds_b128 (ASYNCcnt), weights pre-transposed so both
// A and B tiles are contiguous-row async copies (no in-loop transpose).
// Workspace (80 MB):
//   [0,32MB)   x_norm bf16 [M][D]
//   [32,64MB)  h      bf16 [M][D]
//   [64,72MB)  W1^T   bf16 [n][k]
//   [72,80MB)  W2^T   bf16 [n][k]
// d_out holds x_norm f32 (residual) until GEMM2 accumulates into it.
// ---------------------------------------------------------------------------

typedef __attribute__((ext_vector_type(16))) __bf16 v16bf;
typedef __attribute__((ext_vector_type(8)))  float  v8f;
typedef unsigned short u16;
typedef unsigned int   u32;

#define LN_D    2048
#define LN_M    8192
#define EPSF    1e-6f
#define KSTEP   32
#define NSTEPS  (LN_D / KSTEP)     // 64
#define LPITCH  40                 // LDS row pitch in u16 (80 B, 16B aligned, conflict-free)
#define BUFU16  (256 * LPITCH)     // one double-buffer half: 128 A rows + 128 B rows
#define BUFBYTE (BUFU16 * 2)       // 20480 B

union FragBF { v16bf v; u32 u[8]; };

__device__ __forceinline__ u16 f2bf(float f) {
  u32 u = __builtin_bit_cast(u32, f);
  u32 r = u + 0x7FFFu + ((u >> 16) & 1u);   // round-to-nearest-even
  return (u16)(r >> 16);
}

__device__ __forceinline__ u32 hash32(u32 x) {
  x ^= 0x9E3779B9u;
  x ^= x >> 16; x *= 0x7FEB352Du;
  x ^= x >> 15; x *= 0x846CA68Bu;
  x ^= x >> 16;
  return x;
}

// CDNA5 async DMA: global -> LDS, 16B per lane, tracked by ASYNCcnt.
__device__ __forceinline__ void async_b128(u32 lds_off, const u16* gptr) {
  asm volatile("global_load_async_to_lds_b128 %0, %1, off"
               :: "v"(lds_off), "v"(gptr) : "memory");
}
__device__ __forceinline__ void wait_async0() {
  asm volatile("s_wait_asynccnt 0" ::: "memory");
}

// ---------------- W fp32 [k][n] -> bf16 transposed [n][k] ----------------
__global__ __launch_bounds__(256) void cvt_t_kernel(const float* __restrict__ in,
                                                    u16* __restrict__ outT) {
  __shared__ float tile[32][33];
  const int tx = threadIdx.x, ty = threadIdx.y;
  const int x = blockIdx.x * 32 + tx;     // n
  const int y0 = blockIdx.y * 32;         // k
  #pragma unroll
  for (int j = 0; j < 4; ++j)
    tile[ty + 8 * j][tx] = in[(size_t)(y0 + ty + 8 * j) * LN_D + x];
  __syncthreads();
  const int xo = y0 + tx;                 // k
  const int o0 = blockIdx.x * 32;         // n
  #pragma unroll
  for (int j = 0; j < 4; ++j)
    outT[(size_t)(o0 + ty + 8 * j) * LN_D + xo] = f2bf(tile[tx][ty + 8 * j]);
}

// ---------------- LayerNorm: one row (2048 f32) per block ----------------
__global__ __launch_bounds__(256) void ln_kernel(const float* __restrict__ X,
                                                 float* __restrict__ outF,
                                                 u16* __restrict__ outB) {
  __shared__ float red[256];
  const int t = threadIdx.x;
  const size_t base = (size_t)blockIdx.x * LN_D;
  const float4* xv = (const float4*)(X + base);
  float4 a = xv[t];
  float4 b = xv[t + 256];

  float s = a.x + a.y + a.z + a.w + b.x + b.y + b.z + b.w;
  red[t] = s;
  __syncthreads();
  #pragma unroll
  for (int off = 128; off > 0; off >>= 1) {
    if (t < off) red[t] += red[t + off];
    __syncthreads();
  }
  const float mean = red[0] * (1.0f / (float)LN_D);
  __syncthreads();

  float dx[8] = { a.x - mean, a.y - mean, a.z - mean, a.w - mean,
                  b.x - mean, b.y - mean, b.z - mean, b.w - mean };
  float ss = 0.0f;
  #pragma unroll
  for (int j = 0; j < 8; ++j) ss += dx[j] * dx[j];
  red[t] = ss;
  __syncthreads();
  #pragma unroll
  for (int off = 128; off > 0; off >>= 1) {
    if (t < off) red[t] += red[t + off];
    __syncthreads();
  }
  const float rstd = rsqrtf(red[0] * (1.0f / (float)LN_D) + EPSF);

  float o[8];
  #pragma unroll
  for (int j = 0; j < 8; ++j) o[j] = dx[j] * rstd;

  float4* ov = (float4*)(outF + base);
  ov[t]       = make_float4(o[0], o[1], o[2], o[3]);
  ov[t + 256] = make_float4(o[4], o[5], o[6], o[7]);

  uint2 p0, p1;
  p0.x = (u32)f2bf(o[0]) | ((u32)f2bf(o[1]) << 16);
  p0.y = (u32)f2bf(o[2]) | ((u32)f2bf(o[3]) << 16);
  p1.x = (u32)f2bf(o[4]) | ((u32)f2bf(o[5]) << 16);
  p1.y = (u32)f2bf(o[6]) | ((u32)f2bf(o[7]) << 16);
  *(uint2*)(outB + base + 4 * (size_t)t)         = p0;
  *(uint2*)(outB + base + 4 * (size_t)(t + 256)) = p1;
}

// ---------------- GEMM C[M,D] = A[M,D] x B^T[n][k] via WMMA bf16 ----------
// Block tile 128(M) x 128(N), 256 threads = 8 waves; wave w owns rows
// [w*16,w*16+16) x 128 N = 8 accumulators. K stepped by 32, LDS double-
// buffered, staged by async DMA (4 x b128 per thread per step).
// MODE 0: h = relu(C) -> bf16.  MODE 1: out += dropout(C)*4/3 (residual).
template <int MODE>
__global__ __launch_bounds__(256) void gemm_kernel(const u16* __restrict__ A,
                                                   const u16* __restrict__ BT,
                                                   u16* __restrict__ Hout,
                                                   float* __restrict__ Yout) {
  __shared__ u16 sm[2 * BUFU16];  // [buf][row 0..255][LPITCH]; rows 0-127 = A, 128-255 = B

  const int t    = threadIdx.x;
  const int m0   = blockIdx.y * 128;
  const int n0   = blockIdx.x * 128;
  const int lane = t & 31;
  const int wave = t >> 5;
  const int lm   = lane & 15;     // row (A) or col (B/C) within 16
  const int lh   = lane >> 4;     // lane-half: K/M split

  const u32 smbase = (u32)(size_t)(void*)sm;   // LDS byte offset of sm

  // Per-thread async staging slots: chunks t, t+256, t+512, t+768 of 1024.
  // Chunks [0,512): A tile rows; [512,1024): B tile rows. 16B per chunk.
  const u16* gbase[4];
  u32 loff[4];
  #pragma unroll
  for (int j = 0; j < 4; ++j) {
    const int cid = t + 256 * j;
    const int row = (cid & 511) >> 2;       // 0..127
    const int col = (cid & 3) * 8;          // 0,8,16,24 (u16)
    if (cid < 512) {
      gbase[j] = A  + (size_t)(m0 + row) * LN_D + col;
      loff[j]  = (u32)((row * LPITCH + col) * 2);
    } else {
      gbase[j] = BT + (size_t)(n0 + row) * LN_D + col;
      loff[j]  = (u32)(((128 + row) * LPITCH + col) * 2);
    }
  }

  v8f acc[8] = {};

  // Prologue: stage tile 0 into buffer 0.
  #pragma unroll
  for (int j = 0; j < 4; ++j) async_b128(smbase + loff[j], gbase[j]);
  wait_async0();
  __syncthreads();

  for (int i = 0; i < NSTEPS; ++i) {
    const int buf = i & 1;
    // Kick async DMA for tile i+1 into the other buffer (its last readers
    // finished before the previous barrier).
    if (i + 1 < NSTEPS) {
      const u32 dst = smbase + (u32)((buf ^ 1) * BUFBYTE);
      const int kt  = (i + 1) * KSTEP;
      #pragma unroll
      for (int j = 0; j < 4; ++j) async_b128(dst + loff[j], gbase[j] + kt);
    }

    // Compute tile i.
    const u16* lsA = sm + buf * BUFU16;
    const u16* lsB = lsA + 128 * LPITCH;

    FragBF af;   // A 16x32 bf16 (lh=0: K 0..15; lh=1: K 16..31)
    const u32* aRow = (const u32*)(lsA + (wave * 16 + lm) * LPITCH);
    #pragma unroll
    for (int v = 0; v < 8; ++v) {
      const int k = 2 * v + 16 * lh;
      af.u[v] = aRow[k >> 1];
    }
    #pragma unroll
    for (int nt = 0; nt < 8; ++nt) {
      FragBF bf;   // B 32x16 bf16 (lanes 0-15: K 0..15, 16-31: K 16..31)
      const u32* bCol = (const u32*)(lsB + (nt * 16 + lm) * LPITCH);
      #pragma unroll
      for (int v = 0; v < 8; ++v) bf.u[v] = bCol[v + 8 * lh];
      acc[nt] = __builtin_amdgcn_wmma_f32_16x16x32_bf16(
          false, af.v, false, bf.v, (short)0, acc[nt], false, false);
    }

    if (i + 1 < NSTEPS) wait_async0();   // tile i+1 landed (per-thread)
    __syncthreads();                      // all threads: landed + done reading
  }

  // Epilogue. C layout: lane lm = column, VGPR r -> row r + 8*lh.
  #pragma unroll
  for (int nt = 0; nt < 8; ++nt) {
    const int ncol = n0 + nt * 16 + lm;
    #pragma unroll
    for (int r = 0; r < 8; ++r) {
      const int mrow = m0 + wave * 16 + r + 8 * lh;
      const size_t idx = (size_t)mrow * LN_D + ncol;
      const float y = acc[nt][r];
      if (MODE == 0) {
        Hout[idx] = f2bf(fmaxf(y, 0.0f));
      } else {
        const u32 h = hash32((u32)idx);
        const float yd = (h < 0xC0000000u) ? y * (4.0f / 3.0f) : 0.0f;  // keep p=0.75
        Yout[idx] = Yout[idx] + yd;   // residual x_norm already there
      }
    }
  }
}

extern "C" void kernel_launch(void* const* d_in, const int* in_sizes, int n_in,
                              void* d_out, int out_size, void* d_ws, size_t ws_size,
                              hipStream_t stream) {
  const float* X  = (const float*)d_in[0];   // [4,2048,2048] f32
  const float* W1 = (const float*)d_in[1];   // [2048,2048]   f32
  const float* W2 = (const float*)d_in[2];   // [2048,2048]   f32
  float* out = (float*)d_out;                // [4,2048,2048] f32

  char* ws = (char*)d_ws;
  const size_t MD2 = (size_t)LN_M * LN_D * 2;   // 32 MB
  const size_t DD2 = (size_t)LN_D * LN_D * 2;   //  8 MB
  u16* xnb  = (u16*)(ws);
  u16* hb   = (u16*)(ws + MD2);
  u16* w1bt = (u16*)(ws + 2 * MD2);
  u16* w2bt = (u16*)(ws + 2 * MD2 + DD2);

  // 1) weights -> bf16, transposed to [n][k]
  cvt_t_kernel<<<dim3(64, 64), dim3(32, 8), 0, stream>>>(W1, w1bt);
  cvt_t_kernel<<<dim3(64, 64), dim3(32, 8), 0, stream>>>(W2, w2bt);

  // 2) LayerNorm: x_norm f32 -> d_out (residual), bf16 -> ws
  ln_kernel<<<LN_M, 256, 0, stream>>>(X, out, xnb);

  // 3) h = relu(x_norm @ W1)
  gemm_kernel<0><<<dim3(LN_D / 128, LN_M / 128), 256, 0, stream>>>(xnb, w1bt, hb, nullptr);

  // 4) out = x_norm + dropout(h @ W2)
  gemm_kernel<1><<<dim3(LN_D / 128, LN_M / 128), 256, 0, stream>>>(hb, w2bt, nullptr, out);
}
